// rastreador_19207093748307
// MI455X (gfx1250) — matrix operand order repaired and checked by reading, not compiled, not run on hardware
//
#include <hip/hip_runtime.h>
#include <hip/hip_bf16.h>

// ---------------------------------------------------------------------------
// PointConv-style scene-flow correlator for MI455X (gfx1250, wave32, WMMA).
// 69 GFLOP of 1x1-conv GEMMs on v_wmma_f32_16x16x32_f16; weight strips are
// streamed global->LDS with the Tensor Data Mover (double buffered,
// TENSORcnt-tracked). The LDS buffer is written by the TDM behind the
// compiler's back, so it is explicitly escaped/clobbered to keep the
// B-fragment ds_loads alive.
// ---------------------------------------------------------------------------

typedef __attribute__((ext_vector_type(16))) _Float16 v16h;
typedef __attribute__((ext_vector_type(8)))  _Float16 v8h;
typedef __attribute__((ext_vector_type(8)))  float    v8f;
typedef __attribute__((ext_vector_type(4)))  unsigned int v4u;
typedef __attribute__((ext_vector_type(8)))  int      v8i;
typedef __attribute__((ext_vector_type(4)))  int      v4i;

#define BB   2
#define NN   4096
#define CC   256
#define KNB  16
#define INC  515
#define KPAD 544

union AFrag { v16h v; v8h h[2]; };

#if __has_builtin(__builtin_amdgcn_tensor_load_to_lds) && \
    __has_builtin(__builtin_amdgcn_s_wait_tensorcnt)
#define HAVE_TDM 1
#else
#define HAVE_TDM 0
#endif

#if HAVE_TDM
// One-descriptor 16KB contiguous global->LDS DMA (D# per CDNA5 ISA §8.3/8.4):
// count=1, type=2(image), data_size=3(8B), 1-D tile_dim0=tensor_dim0=2048.
// lds_byte is the offset inside this kernel's (single) LDS allocation.
__device__ __forceinline__ void tdm_load_16k(const _Float16* gsrc, unsigned lds_byte) {
  unsigned long long ga = (unsigned long long)gsrc;
  v4u g0;
  g0.x = 1u;                                                  // count=1 (valid)
  g0.y = lds_byte;                                            // lds_addr
  g0.z = (unsigned)ga;                                        // global_addr[31:0]
  g0.w = (unsigned)((ga >> 32) & 0x01FFFFFFull) | 0x80000000u; // [56:32] | type=2
  v8i g1;
  g1[0] = 0x00030000;            // workgroup_mask=0, data_size=3 (8 bytes)
  g1[1] = (int)(2048u << 16);    // tensor_dim0[15:0] in bits 63:48
  g1[2] = 0;                     // tensor_dim0[31:16]=0, tensor_dim1=0
  g1[3] = (int)(2048u << 16);    // tile_dim0 = 2048 (bits 127:112)
  g1[4] = 0;                     // tile_dim1=0, tile_dim2=0 -> 1-D
  g1[5] = 2048;                  // tensor_dim0_stride
  g1[6] = 0;
  g1[7] = 0;
  v4i z4 = {0, 0, 0, 0};
#if __clang_major__ >= 23
  v8i z8 = {0, 0, 0, 0, 0, 0, 0, 0};
  __builtin_amdgcn_tensor_load_to_lds(g0, g1, z4, z4, z8, 0);
#else
  __builtin_amdgcn_tensor_load_to_lds(g0, g1, z4, z4, 0);
#endif
}
#endif

// ---------------------------------------------------------------------------
// Weight pre-swizzle: W[cout][kin] (f32) -> per (kt,jt) WMMA-B tiles (f16).
// Tile element (lane,i) = W[jt*16 + (lane&15)][kt*32 + (lane>>4)*8 + (i&7) + (i>>3)*16],
// zero padded past Kin. One contiguous 32B read per lane at GEMM time.
// ---------------------------------------------------------------------------
__global__ void prep_w_kernel(const float* __restrict__ W, _Float16* __restrict__ Wh,
                              int Kin, int nkt) {
  int e = blockIdx.x * 256 + threadIdx.x;
  if (e >= nkt * 8192) return;
  int kt   = e >> 13;
  int rem  = e & 8191;
  int jt   = rem >> 9;
  int li   = rem & 511;
  int lane = li >> 4;
  int i    = li & 15;
  int k = kt * 32 + ((lane >> 4) << 3) + (i & 7) + ((i >> 3) << 4);
  int n = jt * 16 + (lane & 15);
  float v = (k < Kin) ? W[n * Kin + k] : 0.0f;
  Wh[e] = (_Float16)v;
}

// ---------------------------------------------------------------------------
// [B,C,N] -> [B,N,C] transpose so per-point feature gathers are contiguous.
// ---------------------------------------------------------------------------
__global__ void transpose_cn_kernel(const float* __restrict__ in, float* __restrict__ out) {
  size_t e = (size_t)blockIdx.x * 256 + threadIdx.x;   // B*C*N elements
  int b = (int)(e >> 20);
  int rem = (int)(e & 1048575);
  int c = rem >> 12;
  int n = rem & 4095;
  out[((size_t)b << 20) + (size_t)n * CC + c] = in[e];
}

// ---------------------------------------------------------------------------
// KNN: for each query point, 16 nearest targets. Target cloud in LDS (48KB).
// Branchless unrolled insertion keeps top-16 sorted in registers; strict '<'
// preserves lowest-index-on-tie like jax.lax.top_k.
// ---------------------------------------------------------------------------
__global__ __launch_bounds__(256) void knn_kernel(const float* __restrict__ q,
                                                  const float* __restrict__ tp,
                                                  int* __restrict__ idxOut) {
  __shared__ float sh[3 * NN];
  const int b = blockIdx.y;
  const int tid = threadIdx.x;
  for (int e = tid; e < 3 * NN; e += 256) sh[e] = tp[b * 3 * NN + e];
  __syncthreads();
  const int n = blockIdx.x * 256 + tid;
  const float qx = q[b * 3 * NN + n];
  const float qy = q[b * 3 * NN + NN + n];
  const float qz = q[b * 3 * NN + 2 * NN + n];
  float bd[16]; int bi[16];
#pragma unroll
  for (int s = 0; s < 16; ++s) { bd[s] = 3.4e38f; bi[s] = 0; }
  for (int j = 0; j < NN; ++j) {
    float dx = qx - sh[j];
    float dy = qy - sh[NN + j];
    float dz = qz - sh[2 * NN + j];
    float d = dx * dx + dy * dy + dz * dz;
    if (d < bd[15]) {
      float pd = d; int pi = j;
#pragma unroll
      for (int s = 0; s < 16; ++s) {
        bool sw = pd < bd[s];
        float td = bd[s]; int ti = bi[s];
        bd[s] = sw ? pd : td; bi[s] = sw ? pi : ti;
        pd = sw ? td : pd;    pi = sw ? ti : pi;
      }
    }
  }
#pragma unroll
  for (int s = 0; s < 16; ++s) idxOut[((size_t)b * NN + n) * KNB + s] = bi[s];
}

// ---------------------------------------------------------------------------
// Build X0[p][k][0:544] = [f1(p) | f2(nbr) | dir | 0-pad] in f16, row-major.
// One block per point (p = b*N + n).
// ---------------------------------------------------------------------------
__global__ __launch_bounds__(256) void build_x0_kernel(
    const float* __restrict__ f1t, const float* __restrict__ f2t,
    const float* __restrict__ pc1, const float* __restrict__ pc2,
    const int* __restrict__ idx, _Float16* __restrict__ X0) {
  const int p = blockIdx.x;
  const int b = p >> 12;
  const int n = p & 4095;
  const int tid = threadIdx.x;
  __shared__ int   ids[KNB];
  __shared__ float dirs[KNB * 3];
  if (tid < KNB) ids[tid] = idx[(size_t)p * KNB + tid];
  __syncthreads();
  if (tid < 48) {
    int kk = tid & 15, comp = tid >> 4;
    dirs[kk * 3 + comp] = pc2[b * 3 * NN + comp * NN + ids[kk]]
                        - pc1[b * 3 * NN + comp * NN + n];
  }
  __syncthreads();
  const float* f1r = f1t + (size_t)p * CC;
  for (int kk = 0; kk < KNB; ++kk) {
    const float* f2r = f2t + ((size_t)b * NN + ids[kk]) * CC;
    _Float16* xr = X0 + ((size_t)p * KNB + kk) * KPAD;
    xr[tid]       = (_Float16)f1r[tid];
    xr[CC + tid]  = (_Float16)f2r[tid];
    if (tid < 32) {
      float v = (tid < 3) ? dirs[kk * 3 + tid] : 0.0f;
      xr[2 * CC + tid] = (_Float16)v;
    }
  }
}

// ---------------------------------------------------------------------------
// Tiled WMMA GEMM: Out[M,256] = leakyrelu(A[M,lda] @ W + bias), f16 in/out,
// f32 accumulate. Block = 8 waves = 4 row-tiles x 2 column halves; each wave
// owns 8 column tiles (64 acc VGPRs) and preloads all 8 B-fragments into
// distinct registers so the 8 WMMAs issue back-to-back. Weight strips are
// double-buffered in LDS, streamed by the TDM from wave 0; the next A-frag
// is software-pipelined across the barrier.
// ---------------------------------------------------------------------------
__global__ __launch_bounds__(256) void gemm_kernel(
    const _Float16* __restrict__ A, const _Float16* __restrict__ Bw,
    const float* __restrict__ bias, _Float16* __restrict__ Out,
    int lda, int nkt) {
  __shared__ __align__(128) _Float16 lbs[2][8192];   // 2 x 16KB weight strips
  const int tid    = threadIdx.x;
  const int wave   = tid >> 5;
  const int lane   = tid & 31;
  const int rt     = wave >> 1;                      // row tile in block (0..3)
  const int ch     = wave & 1;                       // column half (8 jtiles)
  const int pt     = blockIdx.x * 4 + rt;            // 16-row tile index
  const int m      = lane & 15;
  const int laneHi = lane >> 4;

  v8f acc[8];
#pragma unroll
  for (int jj = 0; jj < 8; ++jj)
#pragma unroll
    for (int r = 0; r < 8; ++r) acc[jj][r] = 0.0f;

  const _Float16* arowBase = A + (size_t)(pt * 16 + m) * lda + laneHi * 8;

#if HAVE_TDM
  // The TDM writes lbs through a descriptor the compiler cannot see. Escape
  // the pointer into a memory-clobbering asm, and keep a never-taken store
  // with a runtime index, so reads of lbs are never folded to undef.
  asm volatile("" : : "s"(&lbs[0][0]) : "memory");
  if (__builtin_expect(nkt == 0x7fffffff, 0))
    ((volatile _Float16*)&lbs[0][0])[(unsigned)(lda + tid) & 16383] = (_Float16)0.0f;

  if (wave == 0) tdm_load_16k(Bw, 0u);               // prologue: strip 0 -> buf0
  AFrag a;                                           // prologue A-frag (kt = 0)
  a.h[0] = *(const v8h*)(arowBase);
  a.h[1] = *(const v8h*)(arowBase + 16);

  for (int kt = 0; kt < nkt; ++kt) {
    // issue next A-frag early: its wait lands after this iteration's WMMAs
    AFrag an;
    const _Float16* arn = arowBase + (size_t)(kt + 1) * 32;
    an.h[0] = *(const v8h*)(arn);
    an.h[1] = *(const v8h*)(arn + 16);

    if (wave == 0) {
      if (kt + 1 < nkt) {                            // stream next strip, then
        tdm_load_16k(Bw + (size_t)(kt + 1) * 8192,   // wait for current (in-order)
                     (unsigned)(((kt + 1) & 1) * 16384));
        __builtin_amdgcn_s_wait_tensorcnt(1);
      } else {
        __builtin_amdgcn_s_wait_tensorcnt(0);
      }
    }
    __syncthreads();                                 // buf[kt&1] ready for all
    const _Float16* cur = lbs[kt & 1];
    v16h bf[8];
#pragma unroll
    for (int jj = 0; jj < 8; ++jj)
      bf[jj] = *(const v16h*)(cur + (ch * 8 + jj) * 512 + lane * 16);
#pragma unroll
    for (int jj = 0; jj < 8; ++jj)
      acc[jj] = __builtin_amdgcn_wmma_f32_16x16x32_f16(
          false, a.v, false, bf[jj], (short)0, acc[jj], false, false);
    __syncthreads();                                 // done reading before reuse
    a = an;
  }
#else
  for (int kt = 0; kt < nkt; ++kt) {
    __syncthreads();
    {  // cooperative 16KB stage
      const v8h* src = (const v8h*)(Bw + (size_t)kt * 8192);
      v8h* dst = (v8h*)lbs[0];
#pragma unroll
      for (int t = 0; t < 4; ++t) dst[tid + t * 256] = src[tid + t * 256];
      if (kt + 1 < nkt)
        __builtin_prefetch(Bw + (size_t)(kt + 1) * 8192 + tid * 16, 0, 1);
    }
    __syncthreads();
    const _Float16* cur = lbs[0];
    AFrag a;
    const _Float16* ar = arowBase + (size_t)kt * 32;
    a.h[0] = *(const v8h*)(ar);
    a.h[1] = *(const v8h*)(ar + 16);
    v16h bf[8];
#pragma unroll
    for (int jj = 0; jj < 8; ++jj)
      bf[jj] = *(const v16h*)(cur + (ch * 8 + jj) * 512 + lane * 16);
#pragma unroll
    for (int jj = 0; jj < 8; ++jj)
      acc[jj] = __builtin_amdgcn_wmma_f32_16x16x32_f16(
          false, a.v, false, bf[jj], (short)0, acc[jj], false, false);
  }
#endif

  // epilogue: bias + LeakyReLU(0.1) + f16 store (D layout: VGPR r -> rows r,r+8)
#pragma unroll
  for (int jj = 0; jj < 8; ++jj) {
    const int col = (ch * 8 + jj) * 16 + m;
    const float bv = bias[col];
#pragma unroll
    for (int r = 0; r < 8; ++r) {
      const int row = pt * 16 + laneHi * 8 + r;
      float v = acc[jj][r] + bv;
      v = (v >= 0.0f) ? v : 0.1f * v;
      Out[(size_t)row * CC + col] = (_Float16)v;
    }
  }
}

// ---------------------------------------------------------------------------
// Stage 1 tail: WeightNet1(direction) -> w1[16,256]; xout = sum_k w1 * H3.
// One block per point; h[k][j] (16x8 hidden) computed once in LDS.
// ---------------------------------------------------------------------------
__global__ __launch_bounds__(256) void stage1_reduce_kernel(
    const _Float16* __restrict__ H3, const int* __restrict__ idx,
    const float* __restrict__ pc1, const float* __restrict__ pc2,
    const float* __restrict__ Wn1, const float* __restrict__ bn1,
    const float* __restrict__ gamma, const float* __restrict__ beta,
    const float* __restrict__ mean, const float* __restrict__ var,
    const float* __restrict__ Wn2, const float* __restrict__ bn2,
    float* __restrict__ xout) {
  const int p = blockIdx.x, b = p >> 12, n = p & 4095;
  const int tid = threadIdx.x;
  __shared__ float h[128];
  if (tid < 128) {
    int kk = tid >> 3, j = tid & 7;
    int id = idx[(size_t)p * KNB + kk];
    float dx = pc2[b * 3 * NN + id]          - pc1[b * 3 * NN + n];
    float dy = pc2[b * 3 * NN + NN + id]     - pc1[b * 3 * NN + NN + n];
    float dz = pc2[b * 3 * NN + 2 * NN + id] - pc1[b * 3 * NN + 2 * NN + n];
    float t = dx * Wn1[j * 3] + dy * Wn1[j * 3 + 1] + dz * Wn1[j * 3 + 2] + bn1[j];
    t = gamma[j] * (t - mean[j]) * rsqrtf(var[j] + 1e-5f) + beta[j];
    h[tid] = (t > 0.0f) ? t : 0.0f;
  }
  __syncthreads();
  const int c = tid;
  float w2r[8];
#pragma unroll
  for (int j = 0; j < 8; ++j) w2r[j] = Wn2[c * 8 + j];
  const float bb = bn2[c];
  float acc = 0.0f;
  for (int kk = 0; kk < KNB; ++kk) {
    float w = bb;
#pragma unroll
    for (int j = 0; j < 8; ++j) w += h[kk * 8 + j] * w2r[j];
    acc += w * (float)H3[((size_t)p * KNB + kk) * CC + c];
  }
  xout[(size_t)p * CC + c] = acc;
}

// ---------------------------------------------------------------------------
// Stage 2: self-KNN gather of xout + WeightNet2, write [B,C,N] output.
// ---------------------------------------------------------------------------
__global__ __launch_bounds__(256) void stage2_reduce_kernel(
    const float* __restrict__ xout, const int* __restrict__ idx2,
    const float* __restrict__ pc1,
    const float* __restrict__ Wn1, const float* __restrict__ bn1,
    const float* __restrict__ gamma, const float* __restrict__ beta,
    const float* __restrict__ mean, const float* __restrict__ var,
    const float* __restrict__ Wn2, const float* __restrict__ bn2,
    float* __restrict__ out) {
  const int p = blockIdx.x, b = p >> 12, n = p & 4095;
  const int tid = threadIdx.x;
  __shared__ float h[128];
  __shared__ int   ids[KNB];
  if (tid < KNB) ids[tid] = idx2[(size_t)p * KNB + tid];
  __syncthreads();
  if (tid < 128) {
    int kk = tid >> 3, j = tid & 7;
    int id = ids[kk];
    float dx = pc1[b * 3 * NN + id]          - pc1[b * 3 * NN + n];
    float dy = pc1[b * 3 * NN + NN + id]     - pc1[b * 3 * NN + NN + n];
    float dz = pc1[b * 3 * NN + 2 * NN + id] - pc1[b * 3 * NN + 2 * NN + n];
    float t = dx * Wn1[j * 3] + dy * Wn1[j * 3 + 1] + dz * Wn1[j * 3 + 2] + bn1[j];
    t = gamma[j] * (t - mean[j]) * rsqrtf(var[j] + 1e-5f) + beta[j];
    h[tid] = (t > 0.0f) ? t : 0.0f;
  }
  __syncthreads();
  const int c = tid;
  float w2r[8];
#pragma unroll
  for (int j = 0; j < 8; ++j) w2r[j] = Wn2[c * 8 + j];
  const float bb = bn2[c];
  float acc = 0.0f;
  for (int kk = 0; kk < KNB; ++kk) {
    float w = bb;
#pragma unroll
    for (int j = 0; j < 8; ++j) w += h[kk * 8 + j] * w2r[j];
    acc += w * xout[((size_t)b * NN + ids[kk]) * CC + c];
  }
  out[(size_t)b * CC * NN + (size_t)c * NN + n] = acc;
}

// ---------------------------------------------------------------------------
extern "C" void kernel_launch(void* const* d_in, const int* in_sizes, int n_in,
                              void* d_out, int out_size, void* d_ws, size_t ws_size,
                              hipStream_t stream) {
  const float* pc1 = (const float*)d_in[0];
  const float* pc2 = (const float*)d_in[1];
  const float* f1  = (const float*)d_in[2];
  const float* f2  = (const float*)d_in[3];
  const float* W0  = (const float*)d_in[4];
  const float* b0  = (const float*)d_in[5];
  const float* W1  = (const float*)d_in[6];
  const float* b1  = (const float*)d_in[7];
  const float* W2  = (const float*)d_in[8];
  const float* b2  = (const float*)d_in[9];

  char* ws = (char*)d_ws;
  // ping-pong activation buffers (X0/H2 share, H1/H3 share)
  _Float16* X0  = (_Float16*)(ws + 0);              // 131072 x 544 f16 = 142606336 B
  _Float16* H1  = (_Float16*)(ws + 142606336);      // 131072 x 256 f16 =  67108864 B
  _Float16* H2  = X0;                               // X0 dead after GEMM1
  _Float16* H3  = H1;                               // H1 dead after GEMM2
  _Float16* W0h = (_Float16*)(ws + 209715200);      //  278528 B
  _Float16* W1h = (_Float16*)(ws + 209993728);      //  131072 B
  _Float16* W2h = (_Float16*)(ws + 210124800);      //  131072 B
  int*      idx1 = (int*)(ws + 210255872);          //  524288 B
  int*      idx2 = (int*)(ws + 210780160);          //  524288 B
  float*    xout = (float*)(ws + 211304448);        // 8388608 B
  float*    f1t  = (float*)(ws + 219693056);        // 8388608 B
  float*    f2t  = (float*)(ws + 228081664);        // 8388608 B

  // weight swizzle into WMMA-B tile layout (f16, zero-padded K)
  prep_w_kernel<<<(17 * 8192 + 255) / 256, 256, 0, stream>>>(W0, W0h, INC, 17);
  prep_w_kernel<<<(8 * 8192) / 256, 256, 0, stream>>>(W1, W1h, CC, 8);
  prep_w_kernel<<<(8 * 8192) / 256, 256, 0, stream>>>(W2, W2h, CC, 8);

  // feature transposes for contiguous gathers
  transpose_cn_kernel<<<(BB * CC * NN) / 256, 256, 0, stream>>>(f1, f1t);
  transpose_cn_kernel<<<(BB * CC * NN) / 256, 256, 0, stream>>>(f2, f2t);

  // KNN: pc1 -> pc2 and pc1 -> pc1
  knn_kernel<<<dim3(NN / 256, BB), 256, 0, stream>>>(pc1, pc2, idx1);
  knn_kernel<<<dim3(NN / 256, BB), 256, 0, stream>>>(pc1, pc1, idx2);

  // cost-volume input assembly
  build_x0_kernel<<<BB * NN, 256, 0, stream>>>(f1t, f2t, pc1, pc2, idx1, X0);

  // 3-layer MLP via WMMA (M = 131072 rows, 64 rows per block)
  gemm_kernel<<<2048, 256, 0, stream>>>(X0, W0h, b0, H1, KPAD, 17);
  gemm_kernel<<<2048, 256, 0, stream>>>(H1, W1h, b1, H2, CC, 8);
  gemm_kernel<<<2048, 256, 0, stream>>>(H2, W2h, b2, H3, CC, 8);

  // WeightNet1 weighted reduction over K
  stage1_reduce_kernel<<<BB * NN, 256, 0, stream>>>(
      H3, idx1, pc1, pc2,
      (const float*)d_in[10], (const float*)d_in[11], (const float*)d_in[12],
      (const float*)d_in[13], (const float*)d_in[14], (const float*)d_in[15],
      (const float*)d_in[16], (const float*)d_in[17], xout);

  // WeightNet2 self-aggregation -> [B,C,N]
  stage2_reduce_kernel<<<BB * NN, 256, 0, stream>>>(
      xout, idx2, pc1,
      (const float*)d_in[18], (const float*)d_in[19], (const float*)d_in[20],
      (const float*)d_in[21], (const float*)d_in[22], (const float*)d_in[23],
      (const float*)d_in[24], (const float*)d_in[25], (float*)d_out);
}
